// AttentionInteraction_996432412737
// MI455X (gfx1250) — matrix operand order repaired and checked by reading, not compiled, not run on hardware
//
#include <hip/hip_runtime.h>
#include <math.h>

typedef __bf16 bf16_t;
typedef __attribute__((ext_vector_type(16))) __bf16 v16bf;
typedef __attribute__((ext_vector_type(8)))  __bf16 v8bf;
typedef __attribute__((ext_vector_type(8)))  float  v8f;

#define DMODEL    256
#define NGRAPH    32
#define TILES_MAX 64   // up to 1024 rows per graph (E[block]=256, >40 sigma safe)

__device__ __forceinline__ int imin(int a, int b) { return a < b ? a : b; }

// Build a 16x32 bf16 A-operand register for one lane:
// lane half h holds K [16h..16h+15] of row (lane&15).
__device__ __forceinline__ v16bf loadA(const bf16_t* rowbase, int half) {
    union { v16bf v; v8bf h[2]; } u;
    u.h[0] = *(const v8bf*)(rowbase + 16 * half);
    u.h[1] = *(const v8bf*)(rowbase + 16 * half + 8);
    return u.v;
}

// ---------------- fp32 -> bf16 conversion ----------------
__global__ void cvt_f32_bf16(const float* __restrict__ s, bf16_t* __restrict__ d, int n) {
    int i = blockIdx.x * blockDim.x + threadIdx.x;
    if (i < n) d[i] = (bf16_t)s[i];
}

// ---------------- per-graph block offsets (lower_bound on sorted index) ----------------
__global__ void compute_offsets(const int* __restrict__ idxA, int nA,
                                const int* __restrict__ idxC, int nC,
                                int* __restrict__ offsA, int* __restrict__ offsC) {
    int t = threadIdx.x;
    const int* idx; int n; int* o; int b;
    if (t < 64) { b = t;      if (b > NGRAPH) return; idx = idxA; n = nA; o = offsA; }
    else        { b = t - 64; if (b > NGRAPH) return; idx = idxC; n = nC; o = offsC; }
    int lo = 0, hi = n;
    while (lo < hi) { int mid = (lo + hi) >> 1; if (idx[mid] < b) lo = mid + 1; else hi = mid; }
    o[b] = lo;
}

// ---------------- projection GEMM: P = X @ W^T + bias (NT gemm), bf16 WMMA ----------------
// transposed=0: out[m][n] row-major [N,256] bf16 (for Q and K operands)
// transposed=1: out[n][m] with row stride ostride      (for V -> vT)
__global__ __launch_bounds__(32)
void proj_gemm(const bf16_t* __restrict__ X, const bf16_t* __restrict__ W,
               const float* __restrict__ bias, bf16_t* __restrict__ out,
               int transposed, int ostride, int N) {
    int lane = threadIdx.x, half = lane >> 4, lr = lane & 15;
    int m0 = blockIdx.x * 16, n0 = blockIdx.y * 16;
    int mrow = imin(m0 + lr, N - 1);
    const bf16_t* xrow = X + (size_t)mrow * DMODEL;
    const bf16_t* wrow = W + (size_t)(n0 + lr) * DMODEL;  // B[k,n] = W[n,k]
    v8f acc = {};
#pragma unroll
    for (int s = 0; s < 8; ++s) {
        v16bf a = loadA(xrow + s * 32, half);
        v16bf b = *(const v16bf*)(wrow + s * 32 + 16 * half);
        acc = __builtin_amdgcn_wmma_f32_16x16x32_bf16(false, a, false, b, (short)0, acc, false, false);
    }
    float bn = bias[n0 + lr];
#pragma unroll
    for (int r = 0; r < 8; ++r) {
        int m = m0 + r + 8 * half;                 // C layout: M = r + 8*half, N = lr
        if (m >= N) continue;
        float v = acc[r] + bn;
        if (transposed) out[(size_t)(n0 + lr) * ostride + m] = (bf16_t)v;
        else            out[(size_t)m * DMODEL + n0 + lr]    = (bf16_t)v;
    }
}

// ---------------- column sum of vT: vsum[d] = sum_n vT[d][n] ----------------
__global__ void colsum(const bf16_t* __restrict__ vT, int stride, int nk,
                       float* __restrict__ vsum) {
    __shared__ float red[256];
    const bf16_t* row = vT + (size_t)blockIdx.x * stride;
    float s = 0.f;
    for (int i = threadIdx.x; i < nk; i += 256) s += (float)row[i];
    red[threadIdx.x] = s;
    __syncthreads();
    for (int w = 128; w > 0; w >>= 1) {
        if ((int)threadIdx.x < w) red[threadIdx.x] += red[threadIdx.x + w];
        __syncthreads();
    }
    if (threadIdx.x == 0) vsum[blockIdx.x] = red[0];
}

// ---------------- fused block-diagonal attention ----------------
// out[i] = normalize( h[i] + (sum_{j in blk}(e^{s*scale}-1) v[j] + Vsum) / (sum_blk(e-1) + NK) )
__global__ __launch_bounds__(32)
void attn(const bf16_t* __restrict__ Q, const bf16_t* __restrict__ K,
          const bf16_t* __restrict__ VT, int vstride,
          const float* __restrict__ H, const float* __restrict__ VSUM,
          const int* __restrict__ offsQ, const int* __restrict__ offsK,
          int NK, float* __restrict__ OUT) {
    int b = blockIdx.y;
    int q0 = offsQ[b], q1 = offsQ[b + 1];
    int row0 = q0 + blockIdx.x * 16;
    if (row0 >= q1) return;
    int k0 = offsK[b], k1 = offsK[b + 1];
    int lane = threadIdx.x, half = lane >> 4, lr = lane & 15;
    const float scale = 0.0625f;  // 1/sqrt(256)

    // cache the 16x256 Q tile in A-layout registers (8 k-steps)
    int qrow = imin(row0 + lr, q1 - 1);
    const bf16_t* qp = Q + (size_t)qrow * DMODEL;
    v16bf Aq[8];
#pragma unroll
    for (int s = 0; s < 8; ++s) Aq[s] = loadA(qp + s * 32, half);

    v8f acc[16];
#pragma unroll
    for (int t = 0; t < 16; ++t) acc[t] = (v8f){};
    float zacc[8] = {0, 0, 0, 0, 0, 0, 0, 0};
    __shared__ bf16_t Elds[16 * 32];

    int cbeg = k0 & ~31;  // keep 32B/16B alignment for vector loads
    for (int c = cbeg; c < k1; c += 32) {
        // ---- S = Q @ K^T for two 16-col subtiles ----
        v8f S0 = {}, S1 = {};
        int col0 = imin(c + lr, NK - 1);
        int col1 = imin(c + 16 + lr, NK - 1);
        const bf16_t* kp0 = K + (size_t)col0 * DMODEL + 16 * half;
        const bf16_t* kp1 = K + (size_t)col1 * DMODEL + 16 * half;
        __builtin_prefetch((const void*)(K + (size_t)imin(c + 32 + lr, NK - 1) * DMODEL), 0, 1);
#pragma unroll
        for (int s = 0; s < 8; ++s) {
            v16bf bk = *(const v16bf*)(kp0 + s * 32);
            S0 = __builtin_amdgcn_wmma_f32_16x16x32_bf16(false, Aq[s], false, bk, (short)0, S0, false, false);
        }
#pragma unroll
        for (int s = 0; s < 8; ++s) {
            v16bf bk = *(const v16bf*)(kp1 + s * 32);
            S1 = __builtin_amdgcn_wmma_f32_16x16x32_bf16(false, Aq[s], false, bk, (short)0, S1, false, false);
        }
        // ---- E = exp(S*scale)-1 (masked to block), C-layout -> LDS ----
        bool v0 = (c + lr >= k0) && (c + lr < k1);
        bool v1 = (c + 16 + lr >= k0) && (c + 16 + lr < k1);
#pragma unroll
        for (int r = 0; r < 8; ++r) {
            float e0 = v0 ? (__expf(S0[r] * scale) - 1.0f) : 0.0f;
            float e1 = v1 ? (__expf(S1[r] * scale) - 1.0f) : 0.0f;
            zacc[r] += e0 + e1;
            int m = r + 8 * half;
            Elds[m * 32 + lr]      = (bf16_t)e0;
            Elds[m * 32 + 16 + lr] = (bf16_t)e1;
        }
        // same-wave DS ops are in-order: read E back in A-layout
        union { v16bf v; v8bf h[2]; } ue;
        ue.h[0] = *(const v8bf*)(&Elds[lr * 32 + 16 * half]);
        ue.h[1] = *(const v8bf*)(&Elds[lr * 32 + 16 * half + 8]);
        v16bf Ea = ue.v;
        // ---- acc += E @ V  over all 16 output dim-tiles ----
#pragma unroll
        for (int t = 0; t < 16; ++t) {
            v16bf bv = *(const v16bf*)(VT + (size_t)(t * 16 + lr) * vstride + c + 16 * half);
            acc[t] = __builtin_amdgcn_wmma_f32_16x16x32_bf16(false, Ea, false, bv, (short)0, acc[t], false, false);
        }
    }

    // ---- Z per row: reduce across the 16 lanes of each half ----
    float Z[8];
#pragma unroll
    for (int r = 0; r < 8; ++r) {
        float z = zacc[r];
        z += __shfl_xor(z, 1, 32);
        z += __shfl_xor(z, 2, 32);
        z += __shfl_xor(z, 4, 32);
        z += __shfl_xor(z, 8, 32);
        Z[r] = z + (float)NK;
    }
    // ---- out = h + (acc + Vsum)/Z (in-place in acc) ----
#pragma unroll
    for (int t = 0; t < 16; ++t) {
        float vs = VSUM[t * 16 + lr];
#pragma unroll
        for (int r = 0; r < 8; ++r) {
            int rrow = imin(row0 + r + 8 * half, q1 - 1);
            float hv = H[(size_t)rrow * DMODEL + t * 16 + lr];
            acc[t][r] = hv + (acc[t][r] + vs) / Z[r];
        }
    }
    // ---- L2 normalize per row, then store ----
    float nrm[8];
#pragma unroll
    for (int r = 0; r < 8; ++r) {
        float s = 0.f;
#pragma unroll
        for (int t = 0; t < 16; ++t) { float v = acc[t][r]; s += v * v; }
        s += __shfl_xor(s, 1, 32);
        s += __shfl_xor(s, 2, 32);
        s += __shfl_xor(s, 4, 32);
        s += __shfl_xor(s, 8, 32);
        nrm[r] = 1.0f / fmaxf(sqrtf(s), 1e-12f);
    }
#pragma unroll
    for (int t = 0; t < 16; ++t) {
#pragma unroll
        for (int r = 0; r < 8; ++r) {
            int row = row0 + r + 8 * half;
            if (row < q1) OUT[(size_t)row * DMODEL + t * 16 + lr] = acc[t][r] * nrm[r];
        }
    }
}

// ---------------- host launcher ----------------
extern "C" void kernel_launch(void* const* d_in, const int* in_sizes, int n_in,
                              void* d_out, int out_size, void* d_ws, size_t ws_size,
                              hipStream_t stream) {
    const float* h_ads = (const float*)d_in[0];
    const float* h_cat = (const float*)d_in[1];
    const float* W_f[6] = { (const float*)d_in[2], (const float*)d_in[4], (const float*)d_in[6],
                            (const float*)d_in[8], (const float*)d_in[10], (const float*)d_in[12] };
    const float* bq_ads = (const float*)d_in[3];
    const float* bk_ads = (const float*)d_in[5];
    const float* bv_ads = (const float*)d_in[7];
    const float* bq_cat = (const float*)d_in[9];
    const float* bk_cat = (const float*)d_in[11];
    const float* bv_cat = (const float*)d_in[13];
    const int* idx_ads = (const int*)d_in[14];
    const int* idx_cat = (const int*)d_in[15];

    const int NA = in_sizes[0] / DMODEL;
    const int NC = in_sizes[1] / DMODEL;
    const int vsA = NA + 32, vsC = NC + 32;  // padded vT row strides

    // bump allocator over d_ws (256B aligned chunks)
    char* w = (char*)d_ws;
    auto alloc = [&](size_t bytes) -> char* {
        char* p = w; w += (bytes + 255) & ~(size_t)255; return p;
    };
    bf16_t* hA_bf = (bf16_t*)alloc((size_t)NA * DMODEL * 2);
    bf16_t* hC_bf = (bf16_t*)alloc((size_t)NC * DMODEL * 2);
    bf16_t* Wbf[6];
    for (int i = 0; i < 6; ++i) Wbf[i] = (bf16_t*)alloc((size_t)DMODEL * DMODEL * 2);
    bf16_t* qA  = (bf16_t*)alloc((size_t)NA * DMODEL * 2);
    bf16_t* kA  = (bf16_t*)alloc((size_t)NA * DMODEL * 2);
    bf16_t* qC  = (bf16_t*)alloc((size_t)NC * DMODEL * 2);
    bf16_t* kC  = (bf16_t*)alloc((size_t)NC * DMODEL * 2);
    bf16_t* vTA = (bf16_t*)alloc((size_t)DMODEL * vsA * 2);
    bf16_t* vTC = (bf16_t*)alloc((size_t)DMODEL * vsC * 2);
    float* vsumA = (float*)alloc(DMODEL * 4);
    float* vsumC = (float*)alloc(DMODEL * 4);
    int* offsA = (int*)alloc((NGRAPH + 1) * 4);
    int* offsC = (int*)alloc((NGRAPH + 1) * 4);

    // 1) conversions
    int nA = NA * DMODEL, nC = NC * DMODEL, nW = DMODEL * DMODEL;
    cvt_f32_bf16<<<(nA + 255) / 256, 256, 0, stream>>>(h_ads, hA_bf, nA);
    cvt_f32_bf16<<<(nC + 255) / 256, 256, 0, stream>>>(h_cat, hC_bf, nC);
    for (int i = 0; i < 6; ++i)
        cvt_f32_bf16<<<(nW + 255) / 256, 256, 0, stream>>>(W_f[i], Wbf[i], nW);

    // 2) per-graph offsets
    compute_offsets<<<1, 128, 0, stream>>>(idx_ads, NA, idx_cat, NC, offsA, offsC);

    // 3) projections (WMMA bf16, V stored transposed)
    dim3 pgA((NA + 15) / 16, DMODEL / 16), pgC((NC + 15) / 16, DMODEL / 16);
    proj_gemm<<<pgA, 32, 0, stream>>>(hA_bf, Wbf[0], bq_ads, qA, 0, 0, NA);
    proj_gemm<<<pgA, 32, 0, stream>>>(hA_bf, Wbf[1], bk_ads, kA, 0, 0, NA);
    proj_gemm<<<pgA, 32, 0, stream>>>(hA_bf, Wbf[2], bv_ads, vTA, 1, vsA, NA);
    proj_gemm<<<pgC, 32, 0, stream>>>(hC_bf, Wbf[3], bq_cat, qC, 0, 0, NC);
    proj_gemm<<<pgC, 32, 0, stream>>>(hC_bf, Wbf[4], bk_cat, kC, 0, 0, NC);
    proj_gemm<<<pgC, 32, 0, stream>>>(hC_bf, Wbf[5], bv_cat, vTC, 1, vsC, NC);

    // 4) global V column sums (off-block softmax contribution)
    colsum<<<DMODEL, 256, 0, stream>>>(vTA, vsA, NA, vsumA);
    colsum<<<DMODEL, 256, 0, stream>>>(vTC, vsC, NC, vsumC);

    // 5) fused block-diagonal attention, both directions
    dim3 ag(TILES_MAX, NGRAPH);
    float* outA = (float*)d_out;
    float* outC = (float*)d_out + (size_t)NA * DMODEL;
    attn<<<ag, 32, 0, stream>>>(qA, kC, vTC, vsC, h_ads, vsumC, offsA, offsC, NC, outA);
    attn<<<ag, 32, 0, stream>>>(qC, kA, vTA, vsA, h_cat, vsumA, offsC, offsA, NA, outC);
}
